// GNNClassifier_24945170055619
// MI455X (gfx1250) — compile-verified, matrix operand order checked by reading, not run on hardware
//
#include <hip/hip_runtime.h>

#define D_IN  512
#define D_H   128
#define D_OUT 7
#define EPSBN 1e-5f

typedef __attribute__((ext_vector_type(16))) __bf16 v16bf;
typedef __attribute__((ext_vector_type(8)))  float  v8f;
typedef int v4i __attribute__((vector_size(16)));   // matches builtin param: int4

#if defined(__HIP_DEVICE_COMPILE__) && __has_builtin(__builtin_amdgcn_global_load_async_to_lds_b128)
#define HAVE_ASYNC_LDS 1
#else
#define HAVE_ASYNC_LDS 0
#endif

// ---------------- helpers ----------------
__device__ __forceinline__ unsigned short f2bf(float f) {
    unsigned int u = __float_as_uint(f);
    u += 0x7FFFu + ((u >> 16) & 1u);          // round-to-nearest-even
    return (unsigned short)(u >> 16);
}
__device__ __forceinline__ unsigned int pack2bf(float lo, float hi) {
    return (unsigned int)f2bf(lo) | ((unsigned int)f2bf(hi) << 16);
}
__device__ __forceinline__ float bflo(unsigned int p) { return __uint_as_float(p << 16); }
__device__ __forceinline__ float bfhi(unsigned int p) { return __uint_as_float(p & 0xFFFF0000u); }

// 16-byte global(bf16) -> LDS copy: async (ASYNCcnt-tracked, no VGPR round-trip) when available
__device__ __forceinline__ void g2l_b128(const unsigned short* g, unsigned short* l) {
#if HAVE_ASYNC_LDS
    __builtin_amdgcn_global_load_async_to_lds_b128(
        (__attribute__((address_space(1))) v4i*)(g),
        (__attribute__((address_space(3))) v4i*)(l), 0, 0);
#else
    *(uint4*)l = *(const uint4*)g;
#endif
}
__device__ __forceinline__ void wait_async0() {
#if HAVE_ASYNC_LDS
#if __has_builtin(__builtin_amdgcn_s_wait_asynccnt)
    __builtin_amdgcn_s_wait_asynccnt(0);
#else
    asm volatile("s_wait_asynccnt 0" ::: "memory");
#endif
#endif
}

// ---------------- prep: W[K][128] f32 -> Wt[128][K] bf16 (transposed) ----------------
__global__ void k_w2bf16t(const float* __restrict__ W, unsigned short* __restrict__ Wt, int K) {
    int i = blockIdx.x * blockDim.x + threadIdx.x;
    if (i < K * D_H) {
        int k = i / D_H, n = i % D_H;
        Wt[(size_t)n * K + k] = f2bf(W[i]);
    }
}
// ---------------- prep: f32 array -> bf16 array (4 elems/thread) ----------------
__global__ void k_f2bf_arr(const float* __restrict__ src, unsigned short* __restrict__ dst, int n4) {
    int i = blockIdx.x * blockDim.x + threadIdx.x;
    if (i < n4) {
        float4 f = ((const float4*)src)[i];
        uint2 o; o.x = pack2bf(f.x, f.y); o.y = pack2bf(f.z, f.w);
        ((uint2*)dst)[i] = o;
    }
}

// ---------------- degree / dinv ----------------
__global__ void k_fill1(float* __restrict__ p, int n) {
    int i = blockIdx.x * blockDim.x + threadIdx.x;
    if (i < n) p[i] = 1.0f;                    // self-loop
}
__global__ void k_deg_count(const long long* __restrict__ dst, float* __restrict__ deg, int e) {
    int i = blockIdx.x * blockDim.x + threadIdx.x;
    if (i < e) atomicAdd(&deg[dst[i]], 1.0f);
}
__global__ void k_dinv(float* __restrict__ deg, int n) {
    int i = blockIdx.x * blockDim.x + threadIdx.x;
    if (i < n) deg[i] = rsqrtf(fmaxf(deg[i], 1.0f));
}

// ---------------- WMMA GEMM: C[M][128] = A[M][K] (bf16) @ Wt^T (bf16 [128][K]) ----------------
// Double-buffered LDS; per K-step: wait async fill, 1 barrier, kick next fill, 4 WMMAs/wave.
#define LDT 40   // shorts per LDS row (80B: rows 16B-aligned for b128, banks spread)
__global__ __launch_bounds__(256) void k_gemm_bf16(
    const unsigned short* __restrict__ A,   // [M][K] bf16
    const unsigned short* __restrict__ Wt,  // [128][K] bf16 (N-major)
    float* __restrict__ C, int M, int K)
{
    __shared__ unsigned short lds_a[2][64  * LDT];
    __shared__ unsigned short lds_w[2][128 * LDT];

    const int tid   = threadIdx.x;
    const int lane  = tid & 31;
    const int wave  = tid >> 5;
    const int rtile = wave & 3;     // 4 row tiles of 16
    const int chalf = wave >> 2;    // 2 col halves of 64
    const int m     = lane & 15;
    const int half  = lane >> 4;
    const int row0  = blockIdx.x * 64;

    // staging slots (all 256 threads used for both tiles; 16B per copy)
    const int ar = tid >> 2;              // A row 0..63
    const int ac = (tid & 3) * 8;         // A col (shorts): 0,8,16,24
    const int wn = tid >> 1;              // W row 0..127
    const int wc = (tid & 1) * 16;        // W col (shorts): 0,16
    const bool a_ok = (row0 + ar) < M;    // tail rows: leave garbage, outputs not stored

    v8f acc[4];
#pragma unroll
    for (int i = 0; i < 4; ++i) acc[i] = (v8f)0.0f;

    auto fill = [&](int buf, int k0) {
        if (a_ok) g2l_b128(A + (size_t)(row0 + ar) * K + k0 + ac, &lds_a[buf][ar * LDT + ac]);
        g2l_b128(Wt + (size_t)wn * K + k0 + wc, &lds_w[buf][wn * LDT + wc]);
    };

    const int nk = K >> 5;
    fill(0, 0);
    for (int it = 0; it < nk; ++it) {
        wait_async0();
        __syncthreads();                       // fill(it) visible to all; frags(it-1) all read
        if (it + 1 < nk) fill((it + 1) & 1, (it + 1) * 32);   // overlaps with math below

        const unsigned short* la = lds_a[it & 1];
        const unsigned short* lw = lds_w[it & 1];

        // A fragment (ISA 7.12.2: 16-bit A 16x32, M=lane%16, K pairs split by lane-half)
        union { v16bf v; unsigned int u[8]; } af;
        {
            const unsigned int* arow = (const unsigned int*)&la[(rtile * 16 + m) * LDT];
#pragma unroll
            for (int t = 0; t < 8; ++t) {
                int kk = ((t < 4) ? (2 * t) : (16 + 2 * (t - 4))) + 8 * half;
                af.u[t] = arow[kk >> 1];
            }
        }
#pragma unroll
        for (int ct = 0; ct < 4; ++ct) {
            union { v16bf v; unsigned int u[8]; } bfr;
            const unsigned int* wrow = (const unsigned int*)&lw[(chalf * 64 + ct * 16 + m) * LDT];
#pragma unroll
            for (int t = 0; t < 8; ++t) {
                int kk = ((t < 4) ? (2 * t) : (16 + 2 * (t - 4))) + 8 * half;
                bfr.u[t] = wrow[kk >> 1];
            }
            acc[ct] = __builtin_amdgcn_wmma_f32_16x16x32_bf16(
                false, af.v, false, bfr.v, (short)0, acc[ct], false, false);
        }
    }

    // store: C/D layout row = j + 8*half, col = lane%16
#pragma unroll
    for (int ct = 0; ct < 4; ++ct) {
        const int col = chalf * 64 + ct * 16 + m;
#pragma unroll
        for (int j = 0; j < 8; ++j) {
            int row = row0 + rtile * 16 + j + 8 * half;
            if (row < M) C[(size_t)row * D_H + col] = acc[ct][j];
        }
    }
}

// ---------------- aggregation (f32, L2-resident) ----------------
__global__ void k_agg_self(const float* __restrict__ h, const float* __restrict__ dinv,
                           float* __restrict__ a, int total) {
    int i = blockIdx.x * blockDim.x + threadIdx.x;
    if (i < total) {
        int node = i >> 7;
        float w = dinv[node];
        a[i] = h[i] * w * w;                   // self-loop contribution
    }
}
__global__ __launch_bounds__(256) void k_agg_edges(
    const float* __restrict__ h, const float* __restrict__ dinv,
    const long long* __restrict__ src, const long long* __restrict__ dst,
    float* __restrict__ a, int e)
{
    int g    = blockIdx.x * blockDim.x + threadIdx.x;
    int edge = g >> 5;
    int lane = g & 31;
    if (edge >= e) return;
    long long s = src[edge], d = dst[edge];
    float w = dinv[s] * dinv[d];
    float4 v = ((const float4*)(h + (size_t)s * D_H))[lane];
    float* ad = a + (size_t)d * D_H + lane * 4;
    atomicAdd(ad + 0, v.x * w);
    atomicAdd(ad + 1, v.y * w);
    atomicAdd(ad + 2, v.z * w);
    atomicAdd(ad + 3, v.w * w);
}

// ---------------- bias + BN(eval) + ReLU, f32 agg -> bf16 activation ----------------
__global__ void k_bias_bn_relu_bf16(const float* __restrict__ a, const float* __restrict__ b,
                                    const float* __restrict__ g, const float* __restrict__ be,
                                    const float* __restrict__ rm, const float* __restrict__ rv,
                                    unsigned short* __restrict__ o, int total) {
    int i = blockIdx.x * blockDim.x + threadIdx.x;
    if (i < total) {
        int f = i & (D_H - 1);
        float s = g[f] * rsqrtf(rv[f] + EPSBN);
        float v = (a[i] + b[f] - rm[f]) * s + be[f];
        o[i] = f2bf(v > 0.f ? v : 0.f);
    }
}

// ---------------- FC(128->7) + log_softmax : one wave per node (bf16 input) ----------------
__global__ __launch_bounds__(256) void k_fc_lsm(
    const unsigned short* __restrict__ h, const float* __restrict__ Wfc,
    const float* __restrict__ bfc, float* __restrict__ out, int n)
{
    int g    = blockIdx.x * blockDim.x + threadIdx.x;
    int node = g >> 5;
    int lane = g & 31;
    if (node >= n) return;
    uint2 p = ((const uint2*)(h + (size_t)node * D_H))[lane];   // 4 bf16 per lane
    float h0 = bflo(p.x), h1 = bfhi(p.x), h2 = bflo(p.y), h3 = bfhi(p.y);
    float acc[D_OUT];
#pragma unroll
    for (int j = 0; j < D_OUT; ++j) {
        const float* wr = Wfc + (size_t)(lane * 4) * D_OUT + j;
        acc[j] = h0 * wr[0] + h1 * wr[D_OUT] + h2 * wr[2 * D_OUT] + h3 * wr[3 * D_OUT];
    }
#pragma unroll
    for (int off = 16; off > 0; off >>= 1)
#pragma unroll
        for (int j = 0; j < D_OUT; ++j)
            acc[j] += __shfl_xor(acc[j], off, 32);
    if (lane == 0) {
        float mx = -3.4e38f;
#pragma unroll
        for (int j = 0; j < D_OUT; ++j) { acc[j] += bfc[j]; mx = fmaxf(mx, acc[j]); }
        float sum = 0.f;
#pragma unroll
        for (int j = 0; j < D_OUT; ++j) sum += __expf(acc[j] - mx);
        float lse = mx + __logf(sum);
#pragma unroll
        for (int j = 0; j < D_OUT; ++j) out[(size_t)node * D_OUT + j] = acc[j] - lse;
    }
}

// ---------------- driver ----------------
extern "C" void kernel_launch(void* const* d_in, const int* in_sizes, int n_in,
                              void* d_out, int out_size, void* d_ws, size_t ws_size,
                              hipStream_t stream)
{
    const float*     x   = (const float*)d_in[0];
    const long long* ei  = (const long long*)d_in[1];   // int64 [2][E]
    const float* W1  = (const float*)d_in[2];
    const float* b1  = (const float*)d_in[3];
    const float* g1  = (const float*)d_in[4];
    const float* be1 = (const float*)d_in[5];
    const float* rm1 = (const float*)d_in[6];
    const float* rv1 = (const float*)d_in[7];
    const float* W2  = (const float*)d_in[8];
    const float* b2  = (const float*)d_in[9];
    const float* g2  = (const float*)d_in[10];
    const float* be2 = (const float*)d_in[11];
    const float* rm2 = (const float*)d_in[12];
    const float* rv2 = (const float*)d_in[13];
    const float* Wfc = (const float*)d_in[14];
    const float* bfc = (const float*)d_in[15];

    const int N = in_sizes[0] / D_IN;
    const int E = in_sizes[1] / 2;
    const long long* src = ei;
    const long long* dst = ei + E;

    size_t off = 0;
    auto carve = [&](size_t bytes) -> char* {
        char* p = (char*)d_ws + off;
        off += (bytes + 255) & ~(size_t)255;
        return p;
    };
    float*          dinv = (float*)carve((size_t)N * 4);
    float*          hbuf = (float*)carve((size_t)N * D_H * 4);            // GEMM out (f32)
    float*          abuf = (float*)carve((size_t)N * D_H * 4);            // aggregation (f32 atomics)
    unsigned short* xb   = (unsigned short*)carve((size_t)N * D_IN * 2);  // x in bf16 (L2-resident)
    unsigned short* ab16 = (unsigned short*)carve((size_t)N * D_H * 2);   // activations bf16
    unsigned short* wt1  = (unsigned short*)carve((size_t)D_IN * D_H * 2);
    unsigned short* wt2  = (unsigned short*)carve((size_t)D_H  * D_H * 2);
    (void)ws_size; (void)n_in; (void)out_size;

    const int T  = 256;
    const int NH = N * D_H;

    // prep: weights bf16-transposed, x -> bf16
    k_w2bf16t<<<(D_IN * D_H + T - 1) / T, T, 0, stream>>>(W1, wt1, D_IN);
    k_w2bf16t<<<(D_H  * D_H + T - 1) / T, T, 0, stream>>>(W2, wt2, D_H);
    k_f2bf_arr<<<((size_t)N * D_IN / 4 + T - 1) / T, T, 0, stream>>>(x, xb, N * D_IN / 4);

    // degree -> dinv
    k_fill1<<<(N + T - 1) / T, T, 0, stream>>>(dinv, N);
    k_deg_count<<<(E + T - 1) / T, T, 0, stream>>>(dst, dinv, E);
    k_dinv<<<(N + T - 1) / T, T, 0, stream>>>(dinv, N);

    // ---- layer 1 ----
    k_gemm_bf16<<<(N + 63) / 64, T, 0, stream>>>(xb, wt1, hbuf, N, D_IN);
    k_agg_self<<<(NH + T - 1) / T, T, 0, stream>>>(hbuf, dinv, abuf, NH);
    k_agg_edges<<<((size_t)E * 32 + T - 1) / T, T, 0, stream>>>(hbuf, dinv, src, dst, abuf, E);
    k_bias_bn_relu_bf16<<<(NH + T - 1) / T, T, 0, stream>>>(abuf, b1, g1, be1, rm1, rv1, ab16, NH);

    // ---- layer 2 ----
    k_gemm_bf16<<<(N + 63) / 64, T, 0, stream>>>(ab16, wt2, hbuf, N, D_H);
    k_agg_self<<<(NH + T - 1) / T, T, 0, stream>>>(hbuf, dinv, abuf, NH);
    k_agg_edges<<<((size_t)E * 32 + T - 1) / T, T, 0, stream>>>(hbuf, dinv, src, dst, abuf, E);
    k_bias_bn_relu_bf16<<<(NH + T - 1) / T, T, 0, stream>>>(abuf, b2, g2, be2, rm2, rv2, ab16, NH);

    // ---- head ----
    k_fc_lsm<<<((size_t)N * 32 + T - 1) / T, T, 0, stream>>>(ab16, Wfc, bfc, (float*)d_out, N);
}
